// SAModule_65111704207594
// MI455X (gfx1250) — matrix operand order repaired and checked by reading, not compiled, not run on hardware
//
#include <hip/hip_runtime.h>

typedef __attribute__((ext_vector_type(16))) _Float16 v16h;
typedef __attribute__((ext_vector_type(8)))  _Float16 v8h;
typedef __attribute__((ext_vector_type(8)))  float    v8f;
typedef __attribute__((ext_vector_type(4)))  float    v4f;

#define BB    8
#define NN    4096
#define MM    2048
#define FF    64
#define KK    64          // max neighbors
#define FIN   67
#define FINP  96          // padded to multiple of 32
#define H1D   64
#define H2D   64
#define FOUT  128
#define R2    (0.15f * 0.15f)
#define BIGF  3.0e38f
#define NC    (BB * MM)   // 16384 centroids

// nrt = out_channels/16 (A row tiles), nkc = in_channels/32 (K chunks)
#define NRT1  4
#define NKC1  3
#define NRT2  4
#define NKC2  2
#define NRT3  8
#define NKC3  2

// ---------------- wave32 cross-lane argmax/argmin (tie -> smallest index) ---

__device__ inline void wave_argmax(float& v, int& i) {
#pragma unroll
    for (int mask = 1; mask < 32; mask <<= 1) {
        float ov = __shfl_xor(v, mask, 32);
        int   oi = __shfl_xor(i, mask, 32);
        if (ov > v || (ov == v && oi < i)) { v = ov; i = oi; }
    }
}

__device__ inline void wave_argmin(float& v, int& i) {
#pragma unroll
    for (int mask = 1; mask < 32; mask <<= 1) {
        float ov = __shfl_xor(v, mask, 32);
        int   oi = __shfl_xor(i, mask, 32);
        if (ov < v || (ov == v && oi < i)) { v = ov; i = oi; }
    }
}

// ---------------- Kernel 0: pack W^T into A-fragment lane layout ------------
// A-frag (16x32 f16, this file's contiguous K order): lane<16: row=rt*16+lane,   halves h -> K kc*32 + h
//                                 lane>=16: row=rt*16+lane-16, halves h -> K kc*32 + 16 + h
// packed index: ((rt*nkc + kc)*32 + lane)*16 + h  -> per-lane contiguous 32B.

__device__ inline void pack_one(const float* W, _Float16* Wp, int nrt, int nkc,
                                int kdim_valid, int ncols, int i) {
    int h    = i & 15;
    int t    = i >> 4;
    int lane = t & 31;
    int u    = t >> 5;
    int kc   = u % nkc;
    int rt   = u / nkc;
    int k = kc * 32 + ((lane < 16) ? 0 : 16) + h;
    int n = rt * 16 + (lane & 15);
    float v = (k < kdim_valid) ? W[k * ncols + n] : 0.0f;
    Wp[i] = (_Float16)v;
}

__global__ void packw_kernel(const float* __restrict__ W1, const float* __restrict__ W2,
                             const float* __restrict__ W3,
                             _Float16* __restrict__ W1p, _Float16* __restrict__ W2p,
                             _Float16* __restrict__ W3p) {
    int tid = blockIdx.x * blockDim.x + threadIdx.x;
    int stride = gridDim.x * blockDim.x;
    for (int i = tid; i < NRT1 * NKC1 * 512; i += stride) pack_one(W1, W1p, NRT1, NKC1, FIN, H1D, i);
    for (int i = tid; i < NRT2 * NKC2 * 512; i += stride) pack_one(W2, W2p, NRT2, NKC2, H1D, H2D, i);
    for (int i = tid; i < NRT3 * NKC3 * 512; i += stride) pack_one(W3, W3p, NRT3, NKC3, H2D, FOUT, i);
}

// ---------------- Kernel 1: farthest point sampling (1 block per cloud) -----
// 1024 threads (32 waves), 4 points/thread; wave butterfly + single LDS merge
// per iteration (2 barriers instead of a full tree).

__global__ __launch_bounds__(1024) void fps_kernel(const float* __restrict__ pos,
                                                   int* __restrict__ sel,
                                                   float* __restrict__ q) {
    __shared__ float wrv[32];
    __shared__ int   wri[32];
    __shared__ float lp[3];
    __shared__ int   lastIdx;

    const int b    = blockIdx.x;
    const int tid  = threadIdx.x;
    const int lane = tid & 31;
    const int wave = tid >> 5;
    const float* pc = pos + (size_t)b * NN * 3;

    const int PT = NN / 1024;   // 4 points per thread, contiguous chunk
    float px[PT], py[PT], pz[PT], md[PT];
#pragma unroll
    for (int i = 0; i < PT; ++i) {
        int j = tid * PT + i;
        px[i] = pc[j * 3 + 0]; py[i] = pc[j * 3 + 1]; pz[i] = pc[j * 3 + 2];
        md[i] = BIGF;
    }
    if (tid == 0) lastIdx = 0;
    __syncthreads();

    for (int m = 0; m < MM; ++m) {
        int last = lastIdx;
        if (tid == 0) {
            sel[b * MM + m] = last;
            float lx = pc[last * 3 + 0], ly = pc[last * 3 + 1], lz = pc[last * 3 + 2];
            q[((size_t)b * MM + m) * 3 + 0] = lx;
            q[((size_t)b * MM + m) * 3 + 1] = ly;
            q[((size_t)b * MM + m) * 3 + 2] = lz;
            lp[0] = lx; lp[1] = ly; lp[2] = lz;
        }
        __syncthreads();
        float lx = lp[0], ly = lp[1], lz = lp[2];
        float bv = -1.0f; int bi = 0x7fffffff;
#pragma unroll
        for (int i = 0; i < PT; ++i) {
            float dx = px[i] - lx, dy = py[i] - ly, dz = pz[i] - lz;
            float d  = dx * dx + dy * dy + dz * dz;
            float nm = fminf(md[i], d);
            md[i] = nm;
            int j = tid * PT + i;
            if (nm > bv || (nm == bv && j < bi)) { bv = nm; bi = j; }
        }
        wave_argmax(bv, bi);
        if (lane == 0) { wrv[wave] = bv; wri[wave] = bi; }
        __syncthreads();
        if (wave == 0) {
            float v = wrv[lane]; int i = wri[lane];
            wave_argmax(v, i);
            if (lane == 0) lastIdx = i;
        }
        __syncthreads();
    }
}

// ---------------- Kernel 2: radius search, 64 nearest within r --------------
// 256 threads (8 waves); wave butterfly + 8-entry merge per selection round.

__global__ __launch_bounds__(256) void radius_kernel(const float* __restrict__ pos,
                                                     const int* __restrict__ sel,
                                                     const float* __restrict__ q,
                                                     int* __restrict__ nbr) {
    __shared__ float d2s[NN];
    __shared__ float wrv[8];
    __shared__ int   wri[8];

    const int c    = blockIdx.x;
    const int b    = c / MM;
    const int tid  = threadIdx.x;
    const int lane = tid & 31;
    const int wave = tid >> 5;
    const float* pc = pos + (size_t)b * NN * 3;
    const int selfIdx = sel[c];
    const float qx = q[(size_t)c * 3 + 0];
    const float qy = q[(size_t)c * 3 + 1];
    const float qz = q[(size_t)c * 3 + 2];

    for (int j = tid; j < NN; j += 256) {
        float dx = pc[j * 3 + 0] - qx, dy = pc[j * 3 + 1] - qy, dz = pc[j * 3 + 2] - qz;
        float d = dx * dx + dy * dy + dz * dz;
        d2s[j] = (d <= R2) ? d : BIGF;
    }
    __syncthreads();

    for (int k = 0; k < KK; ++k) {
        float bv = BIGF; int bi = 0x7fffffff;
        for (int j = tid; j < NN; j += 256) {
            float v = d2s[j];
            if (v < bv || (v == bv && j < bi)) { bv = v; bi = j; }
        }
        wave_argmin(bv, bi);
        if (lane == 0) { wrv[wave] = bv; wri[wave] = bi; }
        __syncthreads();
        if (wave == 0) {
            float v  = (lane < 8) ? wrv[lane] : BIGF;
            int   i  = (lane < 8) ? wri[lane] : 0x7fffffff;
            wave_argmin(v, i);
            if (lane == 0) {
                if (v >= BIGF) {
                    nbr[(size_t)c * KK + k] = selfIdx;  // invalid -> duplicate centroid (max-safe)
                } else {
                    nbr[(size_t)c * KK + k] = i;
                    d2s[i] = BIGF;
                }
            }
        }
        __syncthreads();
    }
}

// ---------------- MLP helpers ----------------------------------------------

__device__ inline v16h load_a_packed(const _Float16* __restrict__ Wp, int rt, int kc,
                                     int nkc, int lane) {
    return *(const v16h*)(Wp + ((((rt * nkc + kc) * 32) + lane) << 4));
}

__device__ inline v8h shfl16_v8h(v8h v) {
    union { v8h h; int i[4]; } u;
    u.h = v;
#pragma unroll
    for (int k = 0; k < 4; ++k) u.i[k] = __shfl_xor(u.i[k], 16, 32);
    return u.h;
}

// bias + ReLU + f32->f16 on a D fragment (lane<16: channels rt*16+0..7, lane>=16: rt*16+8..15)
__device__ inline v8h relu_bias_cvt(v8f acc, const float* __restrict__ bias, int rt, int lane) {
    int ch0 = rt * 16 + ((lane < 16) ? 0 : 8);
    v8h r;
#pragma unroll
    for (int v = 0; v < 8; ++v) r[v] = (_Float16)fmaxf(acc[v] + bias[ch0 + v], 0.0f);
    return r;
}

// Assemble next-layer B fragment (K chunk = channels [kc*32, kc*32+32)) from the two
// producing D tiles p_lo = tile 2kc, p_hi = tile 2kc+1, via half-wave exchange.
__device__ inline v16h combine_b_frag(v8h p_lo, v8h p_hi, int lane) {
    v8h x_lo = shfl16_v8h(p_lo);
    v8h x_hi = shfl16_v8h(p_hi);
    v16h r;
    if (lane < 16) {
#pragma unroll
        for (int h = 0; h < 8; ++h) { r[h] = p_lo[h]; r[8 + h] = x_lo[h]; }
    } else {
#pragma unroll
        for (int h = 0; h < 8; ++h) { r[h] = x_hi[h]; r[8 + h] = p_hi[h]; }
    }
    return r;
}

// ---------------- Kernel 3: gather + MLP (WMMA, register-resident) ----------
// Transposed formulation: H^T[outch, nbr] = W^T x F^T. Wave w owns neighbors
// [16w, 16w+16); activations never leave registers between layers.

__global__ __launch_bounds__(128) void mlp_kernel(const float* __restrict__ x,
                                                  const float* __restrict__ pos,
                                                  const float* __restrict__ q,
                                                  const int* __restrict__ nbr,
                                                  const _Float16* __restrict__ W1p,
                                                  const _Float16* __restrict__ W2p,
                                                  const _Float16* __restrict__ W3p,
                                                  const float* __restrict__ b1,
                                                  const float* __restrict__ b2,
                                                  const float* __restrict__ b3,
                                                  float* __restrict__ out) {
    __shared__ float sOut[4 * FOUT];

    const int c    = blockIdx.x;
    const int b    = c / MM;
    const int tid  = threadIdx.x;
    const int lane = tid & 31;
    const int wave = tid >> 5;

    const float* xb = x   + (size_t)b * NN * FF;
    const float* pc = pos + (size_t)b * NN * 3;
    const float qx = q[(size_t)c * 3 + 0];
    const float qy = q[(size_t)c * 3 + 1];
    const float qz = q[(size_t)c * 3 + 2];

    // this lane's neighbor (lane and lane+16 share a neighbor, different K halves)
    const int slot = wave * 16 + (lane & 15);
    const int j    = nbr[(size_t)c * KK + slot];
    const float* xrow = xb + (size_t)j * FF;

    // ---- layer-1 B fragments (F^T), gathered straight into registers ----
    v16h bf1[NKC1];
#pragma unroll
    for (int kc = 0; kc < 2; ++kc) {               // channels 0..63: x features
        int kb = kc * 32 + ((lane < 16) ? 0 : 16);
        const v4f* p = (const v4f*)(xrow + kb);    // 16 consecutive f32, 64B aligned
#pragma unroll
        for (int g = 0; g < 4; ++g) {
            v4f t = p[g];
#pragma unroll
            for (int e = 0; e < 4; ++e) bf1[kc][g * 4 + e] = (_Float16)t[e];
        }
    }
    {                                              // channels 64..95: posdiff + zero pad
        v16h r;
#pragma unroll
        for (int h = 0; h < 16; ++h) r[h] = (_Float16)0.0f;
        if (lane < 16) {
            r[0] = (_Float16)(pc[j * 3 + 0] - qx);
            r[1] = (_Float16)(pc[j * 3 + 1] - qy);
            r[2] = (_Float16)(pc[j * 3 + 2] - qz);
        }
        bf1[2] = r;
    }

    // ---- layer 1: out 64 ch (4 row tiles), K = 96 ----
    v8h p1[NRT1];
#pragma unroll
    for (int rt = 0; rt < NRT1; ++rt) {
        v8f acc;
#pragma unroll
        for (int v = 0; v < 8; ++v) acc[v] = 0.0f;
#pragma unroll
        for (int kc = 0; kc < NKC1; ++kc) {
            v16h a = load_a_packed(W1p, rt, kc, NKC1, lane);
            acc = __builtin_amdgcn_wmma_f32_16x16x32_f16(false, a, false, bf1[kc],
                                                         (short)0, acc, false, false);
        }
        p1[rt] = relu_bias_cvt(acc, b1, rt, lane);
    }

    // ---- layer 2: out 64 ch (4 row tiles), K = 64 ----
    v16h bf2[NKC2];
#pragma unroll
    for (int kc = 0; kc < NKC2; ++kc) bf2[kc] = combine_b_frag(p1[2 * kc], p1[2 * kc + 1], lane);

    v8h p2[NRT2];
#pragma unroll
    for (int rt = 0; rt < NRT2; ++rt) {
        v8f acc;
#pragma unroll
        for (int v = 0; v < 8; ++v) acc[v] = 0.0f;
#pragma unroll
        for (int kc = 0; kc < NKC2; ++kc) {
            v16h a = load_a_packed(W2p, rt, kc, NKC2, lane);
            acc = __builtin_amdgcn_wmma_f32_16x16x32_f16(false, a, false, bf2[kc],
                                                         (short)0, acc, false, false);
        }
        p2[rt] = relu_bias_cvt(acc, b2, rt, lane);
    }

    // ---- layer 3: out 128 ch (8 row tiles), K = 64, + ReLU + max over neighbors ----
    v16h bf3[NKC3];
#pragma unroll
    for (int kc = 0; kc < NKC3; ++kc) bf3[kc] = combine_b_frag(p2[2 * kc], p2[2 * kc + 1], lane);

#pragma unroll
    for (int rt = 0; rt < NRT3; ++rt) {
        v8f acc;
#pragma unroll
        for (int v = 0; v < 8; ++v) acc[v] = 0.0f;
#pragma unroll
        for (int kc = 0; kc < NKC3; ++kc) {
            v16h a = load_a_packed(W3p, rt, kc, NKC3, lane);
            acc = __builtin_amdgcn_wmma_f32_16x16x32_f16(false, a, false, bf3[kc],
                                                         (short)0, acc, false, false);
        }
        // bias + relu (f32), then max over the wave's 16 neighbors (butterfly
        // across lanes; lanes 0-15 and 16-31 hold disjoint channel halves)
        int ch0 = rt * 16 + ((lane < 16) ? 0 : 8);
        float vals[8];
#pragma unroll
        for (int v = 0; v < 8; ++v) vals[v] = fmaxf(acc[v] + b3[ch0 + v], 0.0f);
#pragma unroll
        for (int mask = 1; mask < 16; mask <<= 1)
#pragma unroll
            for (int v = 0; v < 8; ++v) vals[v] = fmaxf(vals[v], __shfl_xor(vals[v], mask, 32));
        if ((lane & 15) == 0) {
#pragma unroll
            for (int v = 0; v < 8; ++v) sOut[wave * FOUT + ch0 + v] = vals[v];
        }
    }
    __syncthreads();

    // cross-wave max (4 waves x 128 channels)
    if (tid < FOUT) {
        float m = sOut[tid];
#pragma unroll
        for (int w = 1; w < 4; ++w) m = fmaxf(m, sOut[w * FOUT + tid]);
        out[(size_t)c * FOUT + tid] = m;
    }
}

// ---------------- Kernel 4: pos_out / batch_out tail ------------------------

__global__ void tail_kernel(const float* __restrict__ q, float* __restrict__ out) {
    int i = blockIdx.x * blockDim.x + threadIdx.x;
    if (i >= NC) return;
    float* pos_out = out + (size_t)NC * FOUT;
    long long* batch_out = (long long*)(pos_out + (size_t)NC * 3);
    pos_out[i * 3 + 0] = q[(size_t)i * 3 + 0];
    pos_out[i * 3 + 1] = q[(size_t)i * 3 + 1];
    pos_out[i * 3 + 2] = q[(size_t)i * 3 + 2];
    batch_out[i] = (long long)(i / MM);
}

// ---------------- launch ----------------------------------------------------

extern "C" void kernel_launch(void* const* d_in, const int* in_sizes, int n_in,
                              void* d_out, int out_size, void* d_ws, size_t ws_size,
                              hipStream_t stream) {
    const float* x   = (const float*)d_in[0];
    const float* pos = (const float*)d_in[1];
    // d_in[2] = batch (int64), unused: layout is dense [B*N]
    const float* W1  = (const float*)d_in[3];
    const float* b1  = (const float*)d_in[4];
    const float* W2  = (const float*)d_in[5];
    const float* b2  = (const float*)d_in[6];
    const float* W3  = (const float*)d_in[7];
    const float* b3  = (const float*)d_in[8];
    float* out = (float*)d_out;

    char* w = (char*)d_ws;
    int*      sel = (int*)w;        w += (size_t)NC * sizeof(int);
    float*    q   = (float*)w;      w += (size_t)NC * 3 * sizeof(float);
    int*      nbr = (int*)w;        w += (size_t)NC * KK * sizeof(int);
    _Float16* W1p = (_Float16*)w;   w += (size_t)NRT1 * NKC1 * 512 * sizeof(_Float16);
    _Float16* W2p = (_Float16*)w;   w += (size_t)NRT2 * NKC2 * 512 * sizeof(_Float16);
    _Float16* W3p = (_Float16*)w;   w += (size_t)NRT3 * NKC3 * 512 * sizeof(_Float16);

    packw_kernel<<<8, 256, 0, stream>>>(W1, W2, W3, W1p, W2p, W3p);
    fps_kernel<<<BB, 1024, 0, stream>>>(pos, sel, q);
    radius_kernel<<<NC, 256, 0, stream>>>(pos, sel, q, nbr);
    mlp_kernel<<<NC, 128, 0, stream>>>(x, pos, q, nbr, W1p, W2p, W3p, b1, b2, b3, out);
    tail_kernel<<<(NC + 255) / 256, 256, 0, stream>>>(q, out);
}
